// DA_HGNN_47081431499244
// MI455X (gfx1250) — compile-verified, matrix operand order checked by reading, not run on hardware
//
#include <hip/hip_runtime.h>
#include <hip/hip_bf16.h>
#include <stdint.h>

// ---------------------------------------------------------------------------
// DA-HGNN forward for MI455X (gfx1250, wave32, WMMA).
//  * density terms cancel in the masked softmax (row-constant shift) -> skipped
//  * H has exactly 11 nz per column -> attention is CSR-sparse (45,056 edges)
//  * heavy GEMMs: v_wmma_f32_16x16x32_bf16, 2x2 register tiling per wave,
//    hot GEMM uses double-buffered global_load_async_to_lds_b128 (ASYNCcnt).
// ---------------------------------------------------------------------------

#define NN      4096
#define DIN     784
#define KP      800        // DIN padded to a multiple of 32 for bf16 WMMA
#define DH      256
#define NHEAD   4
#define DHEAD   64
#define TK      11         // TOP_K + 1 (includes self)
#define LDSP    40         // LDS A-tile stride in bf16 (80 B = 5 x 16 B)
#define BLDSP   72         // LDS B-tile stride for NN gemm (144 B = 9 x 16 B)

static __device__ __constant__ float kDE = 0.30151134457776363f; // 11^-0.5

typedef __bf16 bf16;
typedef __attribute__((ext_vector_type(16))) __bf16 v16bf;
typedef __attribute__((ext_vector_type(8)))  float  v8f;

// K-index inside a bf16 16x32 A/B fragment (CDNA5 ISA 7.12.2 layout).
__device__ __forceinline__ int kmap(int e, int h) {
  int v = e >> 1, s = e & 1;
  return ((v & 4) ? 16 : 0) + h * 8 + ((v & 3) << 1) + s;
}

#define WMMA_BF16(A, B, C) \
  __builtin_amdgcn_wmma_f32_16x16x32_bf16(false, (A), false, (B), (short)0, (C), false, false)

// ---------------------------------------------------------------------------
// Prep: row squared norms + bf16 conversion with zero padding 784 -> 800.
// ---------------------------------------------------------------------------
__global__ void k_prep_x(const float* __restrict__ X, float* __restrict__ sq,
                         bf16* __restrict__ Xb) {
  int i = blockIdx.x, tid = threadIdx.x;
  __shared__ float red[256];
  float acc = 0.f;
  for (int d = tid; d < KP; d += 256) {
    float x = (d < DIN) ? X[i * DIN + d] : 0.f;
    Xb[i * KP + d] = (bf16)x;
    acc += x * x;
  }
  red[tid] = acc;
  __syncthreads();
  for (int s = 128; s > 0; s >>= 1) {
    if (tid < s) red[tid] += red[tid + s];
    __syncthreads();
  }
  if (tid == 0) sq[i] = red[0];
}

__global__ void k_pad_theta(const float* __restrict__ th, bf16* __restrict__ thb) {
  int u = blockIdx.x * 256 + threadIdx.x;
  if (u >= KP * DH) return;
  int r = u / DH, c = u % DH;
  thb[u] = (bf16)((r < DIN) ? th[r * DH + c] : 0.f);
}

__global__ void k_f32_to_bf16(const float* __restrict__ src, bf16* __restrict__ dst, int n) {
  int u = blockIdx.x * 256 + threadIdx.x;
  if (u < n) dst[u] = (bf16)src[u];
}

__global__ void k_zero_i32(int* __restrict__ p, int n) {
  int u = blockIdx.x * 256 + threadIdx.x;
  if (u < n) p[u] = 0;
}

// ---------------------------------------------------------------------------
// dist = |sq_i + sq_j - 2 * X X^T| via bf16 WMMA (NT).
// Block tile 128x64 (8 waves, each wave 32x32 = 2x2 WMMA tiles).
// Double-buffered LDS fed by global_load_async_to_lds_b128 (ASYNCcnt).
// ---------------------------------------------------------------------------
__global__ __launch_bounds__(256) void k_dist_wmma(const bf16* __restrict__ Xb,
                                                   const float* __restrict__ sq,
                                                   float* __restrict__ dist) {
  __shared__ __align__(16) bf16 As[2][128 * LDSP];
  __shared__ __align__(16) bf16 Bs[2][64 * LDSP];
  const int rowBase = blockIdx.y * 128, colBase = blockIdx.x * 64;
  const int tid = threadIdx.x, lane = tid & 31, w = tid >> 5;
  const int wr = (w & 3) * 32, wc = (w >> 2) * 32;   // wave patch origin
  const int mn = lane & 15, h = lane >> 4;

  const v8f vzero = {0.f, 0.f, 0.f, 0.f, 0.f, 0.f, 0.f, 0.f};
  v8f acc00 = vzero, acc01 = vzero, acc10 = vzero, acc11 = vzero;

  // async-copy one 128x32 A-slab + 64x32 B-slab (16B chunks) into buffer `buf`
  auto issue = [&](int buf, int k0) {
    unsigned ldsA = (unsigned)(unsigned long long)(&As[buf][0]);
    unsigned ldsB = (unsigned)(unsigned long long)(&Bs[buf][0]);
#pragma unroll
    for (int uu = 0; uu < 2; ++uu) {                       // 512 A-chunks
      int u = tid + uu * 256;
      int r = u >> 2, c = u & 3;
      unsigned la = ldsA + (unsigned)(r * (LDSP * 2) + c * 16);
      unsigned long long ga =
          (unsigned long long)(const void*)(Xb + (size_t)(rowBase + r) * KP + k0) + c * 16;
      asm volatile("global_load_async_to_lds_b128 %0, %1, off"
                   :: "v"(la), "v"(ga) : "memory");
    }
    {                                                      // 256 B-chunks
      int r = tid >> 2, c = tid & 3;
      unsigned la = ldsB + (unsigned)(r * (LDSP * 2) + c * 16);
      unsigned long long ga =
          (unsigned long long)(const void*)(Xb + (size_t)(colBase + r) * KP + k0) + c * 16;
      asm volatile("global_load_async_to_lds_b128 %0, %1, off"
                   :: "v"(la), "v"(ga) : "memory");
    }
  };

  issue(0, 0);
  asm volatile("s_wait_asynccnt 0" ::: "memory");
  __syncthreads();

  const int nsteps = KP / 32;
  for (int s = 0; s < nsteps; ++s) {
    int cur = s & 1;
    if (s + 1 < nsteps) issue(cur ^ 1, (s + 1) * 32);      // prefetch next slab

    v16bf a0, a1, b0, b1;
#pragma unroll
    for (int e = 0; e < 16; ++e) {
      int kk = kmap(e, h);
      a0[e] = As[cur][(wr + mn) * LDSP + kk];
      a1[e] = As[cur][(wr + 16 + mn) * LDSP + kk];
      b0[e] = Bs[cur][(wc + mn) * LDSP + kk];
      b1[e] = Bs[cur][(wc + 16 + mn) * LDSP + kk];
    }
    acc00 = WMMA_BF16(a0, b0, acc00);
    acc01 = WMMA_BF16(a0, b1, acc01);
    acc10 = WMMA_BF16(a1, b0, acc10);
    acc11 = WMMA_BF16(a1, b1, acc11);

    asm volatile("s_wait_asynccnt 0" ::: "memory");        // next slab landed
    __syncthreads();
  }

#pragma unroll
  for (int r = 0; r < 8; ++r) {
    int m0 = rowBase + wr + r + 8 * h;
    int m1 = m0 + 16;
    int n0 = colBase + wc + mn;
    int n1 = n0 + 16;
    dist[(size_t)m0 * NN + n0] = fabsf(sq[m0] + sq[n0] - 2.f * acc00[r]);
    dist[(size_t)m0 * NN + n1] = fabsf(sq[m0] + sq[n1] - 2.f * acc01[r]);
    dist[(size_t)m1 * NN + n0] = fabsf(sq[m1] + sq[n0] - 2.f * acc10[r]);
    dist[(size_t)m1 * NN + n1] = fabsf(sq[m1] + sq[n1] - 2.f * acc11[r]);
  }
}

// ---------------------------------------------------------------------------
// Generic C = A(MxK) @ B(KxN), bf16 in / f32 out. Block tile 128x64,
// 2x2 WMMA tiles per wave, 64-bit vectorized synchronous staging.
// ---------------------------------------------------------------------------
__global__ __launch_bounds__(256) void k_gemm_nn(const bf16* __restrict__ A,
                                                 const bf16* __restrict__ B,
                                                 float* __restrict__ C,
                                                 int M, int Nv, int K) {
  __shared__ __align__(16) bf16 As[128 * LDSP];
  __shared__ __align__(16) bf16 Bs[32 * BLDSP];
  const int rowBase = blockIdx.y * 128, colBase = blockIdx.x * 64;
  const int tid = threadIdx.x, lane = tid & 31, w = tid >> 5;
  const int wr = (w & 3) * 32, wc = (w >> 2) * 32;
  const int mn = lane & 15, h = lane >> 4;

  const v8f vzero = {0.f, 0.f, 0.f, 0.f, 0.f, 0.f, 0.f, 0.f};
  v8f acc00 = vzero, acc01 = vzero, acc10 = vzero, acc11 = vzero;

  for (int k0 = 0; k0 < K; k0 += 32) {
    if (k0 + 32 < K)
      __builtin_prefetch(&A[(size_t)(rowBase + (tid >> 1)) * K + k0 + 32], 0, 0);
    // A slab: 128 rows x 32 k  = 1024 u64 chunks (4 per thread)
#pragma unroll
    for (int uu = 0; uu < 4; ++uu) {
      int u = tid + uu * 256;
      int r = u >> 3, c = u & 7;
      const unsigned long long* src =
          (const unsigned long long*)(A + (size_t)(rowBase + r) * K + k0);
      *(unsigned long long*)(&As[r * LDSP + c * 4]) = src[c];
    }
    // B slab: 32 k-rows x 64 n = 512 u64 chunks (2 per thread)
#pragma unroll
    for (int uu = 0; uu < 2; ++uu) {
      int u = tid + uu * 256;
      int r = u >> 4, c = u & 15;
      const unsigned long long* src =
          (const unsigned long long*)(B + (size_t)(k0 + r) * Nv + colBase);
      *(unsigned long long*)(&Bs[r * BLDSP + c * 4]) = src[c];
    }
    __syncthreads();

    v16bf a0, a1, b0, b1;
#pragma unroll
    for (int e = 0; e < 16; ++e) {
      int kk = kmap(e, h);
      a0[e] = As[(wr + mn) * LDSP + kk];
      a1[e] = As[(wr + 16 + mn) * LDSP + kk];
      b0[e] = Bs[kk * BLDSP + wc + mn];
      b1[e] = Bs[kk * BLDSP + wc + 16 + mn];
    }
    acc00 = WMMA_BF16(a0, b0, acc00);
    acc01 = WMMA_BF16(a0, b1, acc01);
    acc10 = WMMA_BF16(a1, b0, acc10);
    acc11 = WMMA_BF16(a1, b1, acc11);
    __syncthreads();
  }

#pragma unroll
  for (int r = 0; r < 8; ++r) {
    int m0 = rowBase + wr + r + 8 * h;
    int m1 = m0 + 16;
    int n0 = colBase + wc + mn;
    int n1 = n0 + 16;
    C[(size_t)m0 * Nv + n0] = acc00[r];
    C[(size_t)m0 * Nv + n1] = acc01[r];
    C[(size_t)m1 * Nv + n0] = acc10[r];
    C[(size_t)m1 * Nv + n1] = acc11[r];
  }
}

// ---------------------------------------------------------------------------
// Per-row 11 smallest distances (argmin x11 over a 16KB LDS-cached row).
// ---------------------------------------------------------------------------
__global__ __launch_bounds__(256) void k_topk(const float* __restrict__ dist,
                                              int* __restrict__ idx) {
  __shared__ float v[NN];
  __shared__ float rv[256];
  __shared__ int   ri[256];
  int i = blockIdx.x, tid = threadIdx.x;
  for (int u = tid; u < NN; u += 256) v[u] = dist[(size_t)i * NN + u];
  __syncthreads();
  for (int t = 0; t < TK; ++t) {
    float bv = 3.4e38f;
    int bi = 0;
    for (int u = tid; u < NN; u += 256)
      if (v[u] < bv) { bv = v[u]; bi = u; }
    rv[tid] = bv; ri[tid] = bi;
    __syncthreads();
    for (int s = 128; s > 0; s >>= 1) {
      if (tid < s && rv[tid + s] < rv[tid]) { rv[tid] = rv[tid + s]; ri[tid] = ri[tid + s]; }
      __syncthreads();
    }
    if (tid == 0) { idx[i * TK + t] = ri[0]; v[ri[0]] = 3.4e38f; }
    __syncthreads();
  }
}

// ---------------------------------------------------------------------------
// Degrees + CSR of H's rows. Column j of H = idx[j][0..10].
// ---------------------------------------------------------------------------
__global__ void k_count(const int* __restrict__ idx, int* __restrict__ cnt) {
  int u = blockIdx.x * 256 + threadIdx.x;
  if (u < NN * TK) atomicAdd(&cnt[idx[u]], 1);
}

__global__ void k_dv(const int* __restrict__ cnt, float* __restrict__ dv) {
  int u = blockIdx.x * 256 + threadIdx.x;
  if (u < NN) dv[u] = rsqrtf((float)max(cnt[u], 1));
}

__global__ __launch_bounds__(1024) void k_scan(const int* __restrict__ cnt,
                                               int* __restrict__ rowptr) {
  __shared__ int sh[1024];
  int tid = threadIdx.x;
  int c0 = cnt[4 * tid], c1 = cnt[4 * tid + 1], c2 = cnt[4 * tid + 2], c3 = cnt[4 * tid + 3];
  int part = c0 + c1 + c2 + c3;
  sh[tid] = part;
  __syncthreads();
  for (int off = 1; off < 1024; off <<= 1) {
    int v2 = (tid >= off) ? sh[tid - off] : 0;
    __syncthreads();
    sh[tid] += v2;
    __syncthreads();
  }
  int run = sh[tid] - part;
  rowptr[4 * tid + 0] = run; run += c0;
  rowptr[4 * tid + 1] = run; run += c1;
  rowptr[4 * tid + 2] = run; run += c2;
  rowptr[4 * tid + 3] = run; run += c3;
  if (tid == 1023) rowptr[NN] = run;
}

__global__ void k_fill(const int* __restrict__ idx, const int* __restrict__ rowptr,
                       int* __restrict__ fill, int* __restrict__ colidx) {
  int u = blockIdx.x * 256 + threadIdx.x;
  if (u >= NN * TK) return;
  int j = u / TK;
  int i = idx[u];
  int pos = atomicAdd(&fill[i], 1);
  colidx[rowptr[i] + pos] = j;
}

// G[j] = de * sum_t dv[p_t] * X[p_t]  (bf16, zero-padded to KP)
__global__ void k_gather(const float* __restrict__ X, const int* __restrict__ idx,
                         const float* __restrict__ dv, bf16* __restrict__ Gb) {
  int j = blockIdx.x, tid = threadIdx.x;
  __shared__ int   nb[TK];
  __shared__ float dvw[TK];
  if (tid < TK) { int p = idx[j * TK + tid]; nb[tid] = p; dvw[tid] = dv[p]; }
  __syncthreads();
  for (int d = tid; d < KP; d += 256) {
    float acc = 0.f;
    if (d < DIN) {
#pragma unroll
      for (int t = 0; t < TK; ++t) acc += dvw[t] * X[(size_t)nb[t] * DIN + d];
    }
    Gb[(size_t)j * KP + d] = (bf16)(kDE * acc);
  }
}

// E[i] = dv_i * de * sum_{j in CSR row i} Xe[j]
__global__ void k_E(const float* __restrict__ Xe, const int* __restrict__ rowptr,
                    const int* __restrict__ colidx, const float* __restrict__ dv,
                    float* __restrict__ E, bf16* __restrict__ Eb) {
  int i = blockIdx.x, d = threadIdx.x;
  int beg = rowptr[i], end = rowptr[i + 1];
  float acc = 0.f;
  for (int e = beg; e < end; ++e) acc += Xe[(size_t)colidx[e] * DH + d];
  float val = dv[i] * kDE * acc;
  E[(size_t)i * DH + d] = val;
  Eb[(size_t)i * DH + d] = (bf16)val;
}

// out[r] = Mt[r,:ld] . vec
__global__ void k_matvec(const float* __restrict__ Mt, int ld,
                         const float* __restrict__ vec, float* __restrict__ out) {
  int r = blockIdx.x * 256 + threadIdx.x;
  if (r >= NN) return;
  float acc = 0.f;
  for (int d = 0; d < ld; ++d) acc += Mt[(size_t)r * ld + d] * vec[d];
  out[r] = acc;
}

// CSR masked softmax-attention + ELU. rho terms cancel (row-constant shift).
__global__ void k_att(const int* __restrict__ rowptr, const int* __restrict__ colidx,
                      const float* __restrict__ s, const float* __restrict__ t,
                      const float* __restrict__ val, int ldv,
                      float* __restrict__ out, int ldo, int off) {
  int i = blockIdx.x, d = threadIdx.x;
  int beg = rowptr[i], end = rowptr[i + 1];
  if (beg == end) { out[(size_t)i * ldo + off + d] = 0.f; return; }
  float si = s[i];
  float mx = -3.4e38f;
  for (int e = beg; e < end; ++e) {
    float a = si + t[colidx[e]];
    a = (a > 0.f) ? a : 0.2f * a;        // leaky_relu(0.2)
    mx = fmaxf(mx, a);
  }
  float acc = 0.f, wsum = 0.f;
  for (int e = beg; e < end; ++e) {
    int j = colidx[e];
    float a = si + t[j];
    a = (a > 0.f) ? a : 0.2f * a;
    float wgt = __expf(a - mx);
    wsum += wgt;
    acc += wgt * val[(size_t)j * ldv + d];
  }
  float o = acc / wsum;
  out[(size_t)i * ldo + off + d] = (o > 0.f) ? o : (__expf(o) - 1.f); // elu
}

// ---------------------------------------------------------------------------
// Host-side orchestration.
// ---------------------------------------------------------------------------
extern "C" void kernel_launch(void* const* d_in, const int* in_sizes, int n_in,
                              void* d_out, int out_size, void* d_ws, size_t ws_size,
                              hipStream_t stream) {
  (void)in_sizes; (void)n_in; (void)out_size; (void)ws_size;
  const float* X       = (const float*)d_in[0];
  const float* theta   = (const float*)d_in[1];
  const float* W_heads = (const float*)d_in[2];
  const float* ax_h    = (const float*)d_in[3];
  const float* ae_h    = (const float*)d_in[4];
  const float* W2      = (const float*)d_in[5];
  const float* ax2     = (const float*)d_in[6];
  const float* ae2     = (const float*)d_in[7];
  float* out = (float*)d_out;

  uint8_t* base = (uint8_t*)d_ws;
  size_t off = 0;
  auto alloc = [&](size_t bytes) -> void* {
    void* r = base + off;
    off = (off + bytes + 255) & ~(size_t)255;
    return r;
  };

  float* sq      = (float*)alloc(NN * 4);
  bf16*  Xb      = (bf16*) alloc((size_t)NN * KP * 2);
  float* dist    = (float*)alloc((size_t)NN * NN * 4);
  int*   idx     = (int*)  alloc(NN * TK * 4);
  int*   cnt     = (int*)  alloc(NN * 4);
  float* dv      = (float*)alloc(NN * 4);
  int*   rowptr  = (int*)  alloc((NN + 1) * 4);
  int*   fill    = (int*)  alloc(NN * 4);
  int*   colidx  = (int*)  alloc(NN * TK * 4);
  bf16*  thetab  = (bf16*) alloc((size_t)KP * DH * 2);
  bf16*  Gb      = (bf16*) alloc((size_t)NN * KP * 2);
  float* Xe      = (float*)alloc((size_t)NN * DH * 4);
  bf16*  Xeb     = (bf16*) alloc((size_t)NN * DH * 2);
  float* E       = (float*)alloc((size_t)NN * DH * 4);
  bf16*  Eb      = (bf16*) alloc((size_t)NN * DH * 2);
  bf16*  Wb      = (bf16*) alloc((size_t)NHEAD * DH * DHEAD * 2);
  bf16*  W2b     = (bf16*) alloc((size_t)DH * DH * 2);
  float* WX      = (float*)alloc((size_t)NN * DHEAD * 4);
  float* WE      = (float*)alloc((size_t)NN * DHEAD * 4);
  float* sbuf    = (float*)alloc(NN * 4);
  float* tbuf    = (float*)alloc(NN * 4);
  float* Ehead   = (float*)alloc((size_t)NN * DHEAD * 4);
  float* Xcat    = (float*)alloc((size_t)NN * DH * 4);
  bf16*  Xcb     = (bf16*) alloc((size_t)NN * DH * 2);
  float* WXc     = (float*)alloc((size_t)NN * DH * 4);
  float* WEc     = (float*)alloc((size_t)NN * DH * 4);
  float* sa      = (float*)alloc(NN * 4);
  float* ta      = (float*)alloc(NN * 4);
  float* sb      = (float*)alloc(NN * 4);
  float* tb      = (float*)alloc(NN * 4);
  float* E2      = (float*)alloc((size_t)NN * DH * 4);

  // --- hypergraph construction -------------------------------------------
  k_prep_x<<<NN, 256, 0, stream>>>(X, sq, Xb);
  k_pad_theta<<<(KP * DH + 255) / 256, 256, 0, stream>>>(theta, thetab);
  k_dist_wmma<<<dim3(NN / 64, NN / 128), 256, 0, stream>>>(Xb, sq, dist);
  k_topk<<<NN, 256, 0, stream>>>(dist, idx);
  k_zero_i32<<<NN / 256, 256, 0, stream>>>(cnt, NN);
  k_zero_i32<<<NN / 256, 256, 0, stream>>>(fill, NN);
  k_count<<<(NN * TK + 255) / 256, 256, 0, stream>>>(idx, cnt);
  k_dv<<<NN / 256, 256, 0, stream>>>(cnt, dv);
  k_scan<<<1, 1024, 0, stream>>>(cnt, rowptr);
  k_fill<<<(NN * TK + 255) / 256, 256, 0, stream>>>(idx, rowptr, fill, colidx);

  // --- HGCN layer: Xe = De H^T Dv X theta ; E = Dv H De Xe ----------------
  k_gather<<<NN, 256, 0, stream>>>(X, idx, dv, Gb);
  k_gemm_nn<<<dim3(DH / 64, NN / 128), 256, 0, stream>>>(Gb, thetab, Xe, NN, DH, KP);
  k_f32_to_bf16<<<(NN * DH + 255) / 256, 256, 0, stream>>>(Xe, Xeb, NN * DH);
  k_E<<<NN, DH, 0, stream>>>(Xe, rowptr, colidx, dv, E, Eb);

  // --- weight conversions -------------------------------------------------
  k_f32_to_bf16<<<(NHEAD * DH * DHEAD + 255) / 256, 256, 0, stream>>>(W_heads, Wb, NHEAD * DH * DHEAD);
  k_f32_to_bf16<<<(DH * DH + 255) / 256, 256, 0, stream>>>(W2, W2b, DH * DH);

  // --- multi-head DA-HGAN (density terms cancel in softmax) ---------------
  for (int h = 0; h < NHEAD; ++h) {
    const bf16*  Wh  = Wb + (size_t)h * DH * DHEAD;
    const float* axh = ax_h + (size_t)h * 2 * DHEAD;
    const float* aeh = ae_h + (size_t)h * 2 * DHEAD;
    k_gemm_nn<<<dim3(DHEAD / 64, NN / 128), 256, 0, stream>>>(Xeb, Wh, WX, NN, DHEAD, DH);
    k_gemm_nn<<<dim3(DHEAD / 64, NN / 128), 256, 0, stream>>>(Eb,  Wh, WE, NN, DHEAD, DH);
    // E_new = elu( softmax_H( lrelu(s_i + t_j) ) @ (Xe W) )
    k_matvec<<<NN / 256, 256, 0, stream>>>(WX, DHEAD, axh,         sbuf);
    k_matvec<<<NN / 256, 256, 0, stream>>>(WE, DHEAD, axh + DHEAD, tbuf);
    k_att<<<NN, DHEAD, 0, stream>>>(rowptr, colidx, sbuf, tbuf, WX, DHEAD, Ehead, DHEAD, 0);
    // X_new = elu( softmax_H( lrelu(s2_i + t2_j) ) @ E_new ), into head slice
    k_matvec<<<NN / 256, 256, 0, stream>>>(WE, DHEAD, aeh,         sbuf);
    k_matvec<<<NN / 256, 256, 0, stream>>>(WX, DHEAD, aeh + DHEAD, tbuf);
    k_att<<<NN, DHEAD, 0, stream>>>(rowptr, colidx, sbuf, tbuf, Ehead, DHEAD, Xcat, DH, h * DHEAD);
  }

  // --- level-2 DA-HGAN ----------------------------------------------------
  k_f32_to_bf16<<<(NN * DH + 255) / 256, 256, 0, stream>>>(Xcat, Xcb, NN * DH);
  k_gemm_nn<<<dim3(DH / 64, NN / 128), 256, 0, stream>>>(Xcb, W2b, WXc, NN, DH, DH);
  k_gemm_nn<<<dim3(DH / 64, NN / 128), 256, 0, stream>>>(Eb,  W2b, WEc, NN, DH, DH);

  k_matvec<<<NN / 256, 256, 0, stream>>>(WXc, DH, ax2,      sa);
  k_matvec<<<NN / 256, 256, 0, stream>>>(WEc, DH, ax2 + DH, ta);
  k_att<<<NN, DH, 0, stream>>>(rowptr, colidx, sa, ta, WXc, DH, E2, DH, 0);

  k_matvec<<<NN / 256, 256, 0, stream>>>(WEc, DH, ae2,      sb);
  k_matvec<<<NN / 256, 256, 0, stream>>>(WXc, DH, ae2 + DH, tb);
  k_att<<<NN, DH, 0, stream>>>(rowptr, colidx, sb, tb, E2, DH, out, DH, 0);
}
